// T_att_real_1803886264320
// MI455X (gfx1250) — matrix operand order repaired and checked by reading, not compiled, not run on hardware
//
#include <hip/hip_runtime.h>
#include <hip/hip_bf16.h>

typedef __attribute__((ext_vector_type(16))) __bf16 v16bf;
typedef __attribute__((ext_vector_type(8)))  __bf16 v8bf;
typedef __attribute__((ext_vector_type(8)))  float  v8f;
typedef __attribute__((ext_vector_type(4)))  unsigned int u32x4;
typedef __attribute__((ext_vector_type(8)))  int  i32x8;
typedef __attribute__((ext_vector_type(4)))  int  i32x4;

#define NBATCH 8
#define SEQ    2048
#define INCH   512
#define HIDN   512
#define SCALE_F 0.25f   // 1/sqrt(16)

// ---------------------------------------------------------------------------
// Kernel 0: transpose + convert weights to bf16.  Wt[p][n*512+k] = W_p[k*512+n]
// ---------------------------------------------------------------------------
__global__ __launch_bounds__(256) void wtrans_kernel(const float* __restrict__ Wq,
                                                     const float* __restrict__ Wk,
                                                     const float* __restrict__ Wv,
                                                     __bf16* __restrict__ Wt) {
  int idx = blockIdx.x * 256 + threadIdx.x;       // 3*512*512 total
  int p = idx >> 18;
  int r = idx & ((1 << 18) - 1);
  int n = r >> 9;
  int k = r & 511;
  const float* W = (p == 0) ? Wq : (p == 1) ? Wk : Wv;
  Wt[idx] = (__bf16)W[k * 512 + n];
}

// ---------------------------------------------------------------------------
// Kernel 1: QKV projection GEMM.  One wave computes a 16x64 output tile:
// one fp32->bf16 A-fragment feeds 4 WMMAs (4x data reuse on the fp32 input).
//   proj 0 -> Qp[m][n] (bf16, row major)
//   proj 1 -> Kp[m][n] (bf16, row major)
//   proj 2 -> Vt[b][n][t] (bf16, transposed per batch for PV B-fragments)
// ---------------------------------------------------------------------------
__global__ __launch_bounds__(256) void qkv_proj_kernel(
    const float* __restrict__ q, const float* __restrict__ k,
    const float* __restrict__ v, const float* __restrict__ bq,
    const float* __restrict__ bk, const float* __restrict__ bv,
    const __bf16* __restrict__ Wt, __bf16* __restrict__ Qp,
    __bf16* __restrict__ Kp, __bf16* __restrict__ Vt) {
  const int wid  = blockIdx.x * 8 + (threadIdx.x >> 5);
  const int lane = threadIdx.x & 31;
  const int hi   = lane >> 4;       // high half of wave
  const int l15  = lane & 15;

  // 3 proj * 1024 m-tiles * 8 n-groups (64 wide) = 24576 waves
  const int proj = wid >> 13;                  // / 8192
  const int rem  = wid & 8191;
  const int mt   = rem >> 3;
  const int ng   = rem & 7;
  const int m0   = mt * 16, n0 = ng * 64;

  const float*  X    = (proj == 0) ? q  : (proj == 1) ? k  : v;
  const float*  bias = (proj == 0) ? bq : (proj == 1) ? bk : bv;
  const __bf16* W    = Wt + proj * (512 * 512);

  const int    arow = m0 + l15;           // A-matrix row for this lane
  const float* arp  = X + (size_t)arow * 512;

  v8f acc[4];
#pragma unroll
  for (int j = 0; j < 4; ++j)
#pragma unroll
    for (int i = 0; i < 8; ++i) acc[j][i] = 0.0f;

  for (int k0 = 0; k0 < 512; k0 += 32) {
    // A fragment: fp32 -> bf16 in-register.  Lane holds K = {k0+hi*16..+15} (contiguous)
    v8f g1 = *(const v8f*)(arp + k0 + hi * 16);
    v8f g2 = *(const v8f*)(arp + k0 + hi * 16 + 8);
    v16bf a;
#pragma unroll
    for (int i = 0; i < 8; ++i) { a[i] = (__bf16)g1[i]; a[i + 8] = (__bf16)g2[i]; }
#pragma unroll
    for (int j = 0; j < 4; ++j) {
      // B fragment: contiguous 16 K-values from transposed weights, col n0+j*16+l15
      const __bf16* wcp = W + (size_t)(n0 + j * 16 + l15) * 512 + k0 + hi * 16;
      v8bf b1 = *(const v8bf*)(wcp);
      v8bf b2 = *(const v8bf*)(wcp + 8);
      v16bf bfr;
#pragma unroll
      for (int i = 0; i < 8; ++i) { bfr[i] = b1[i]; bfr[i + 8] = b2[i]; }
      acc[j] = __builtin_amdgcn_wmma_f32_16x16x32_bf16(false, a, false, bfr,
                                                       (short)0, acc[j], false, false);
    }
  }

  if (proj < 2) {
    __bf16* P = (proj == 0) ? Qp : Kp;
#pragma unroll
    for (int j = 0; j < 4; ++j) {
      const int   ncol = n0 + j * 16 + l15;
      const float bb   = bias[ncol];
#pragma unroll
      for (int i = 0; i < 8; ++i) {
        int m = m0 + i + hi * 8;
        P[(size_t)m * 512 + ncol] = (__bf16)(acc[j][i] + bb);
      }
    }
  } else {
#pragma unroll
    for (int j = 0; j < 4; ++j) {
      const int   ncol = n0 + j * 16 + l15;
      const float bb   = bias[ncol];
#pragma unroll
      for (int i = 0; i < 8; ++i) {
        int m = m0 + i + hi * 8;
        int bnum = m >> 11, t = m & 2047;
        Vt[((size_t)(bnum * 512 + ncol)) * 2048 + t] = (__bf16)(acc[j][i] + bb);
      }
    }
  }
}

// ---------------------------------------------------------------------------
// TDM: stage one 32x512 bf16 K block (32 KB) into LDS with a single
// tensor_load_to_lds.  D# built per CDNA5 ISA.
// amdgpu-toolchain (clang-23) 6-arg builtin form.
// ---------------------------------------------------------------------------
__device__ __forceinline__ void tdm_load_kblock(const __bf16* gsrc,
                                                __bf16* lds_dst) {
  unsigned lds_off = (unsigned)(size_t)lds_dst;        // low 32 bits = LDS byte offset
  unsigned long long ga = (unsigned long long)(size_t)gsrc;
  u32x4 g0;
  g0[0] = 1u;                                          // count=1, is_restore=0
  g0[1] = lds_off;                                     // lds_addr
  g0[2] = (unsigned)ga;                                // global_addr[31:0]
  g0[3] = (unsigned)((ga >> 32) & 0x01FFFFFFu) | (2u << 30);  // addr[56:32] | type=2
  i32x8 g1;
  g1[0] = 0x00010000;            // workgroup_mask=0, data_size=1 (2 bytes)
  g1[1] = (int)(512u << 16);     // tensor_dim0 low16 = 512 (row length, elements)
  g1[2] = (int)(32u << 16);      // tensor_dim0 hi=0 | tensor_dim1 low16 = 32 rows
  g1[3] = (int)(512u << 16);     // tensor_dim1 hi=0 | tile_dim0 = 512
  g1[4] = 32;                    // tile_dim1 = 32, tile_dim2 = 0
  g1[5] = 512;                   // tensor_dim0_stride low32 = 512 elements
  g1[6] = 0;                     // dim0_stride hi | dim1_stride low = 0
  g1[7] = 0;
  i32x4 gz4;
  gz4[0] = 0; gz4[1] = 0; gz4[2] = 0; gz4[3] = 0;      // groups 2/3 unused (2-D)
  i32x8 gz8;
#pragma unroll
  for (int i = 0; i < 8; ++i) gz8[i] = 0;
  __builtin_amdgcn_tensor_load_to_lds(g0, g1, gz4, gz4, gz8, 0);
}

// ---------------------------------------------------------------------------
// Kernel 2: flash attention.  One block (8 waves) per 16 query rows.
// Wave w owns H-chunk [w*64, w*64+64).  Online softmax, replicated per wave.
// K blocks staged via TDM into LDS, pipelined across the block loop.
// ---------------------------------------------------------------------------
__global__ __launch_bounds__(256) void attn_kernel(const __bf16* __restrict__ Qp,
                                                   const __bf16* __restrict__ Kp,
                                                   const __bf16* __restrict__ Vt,
                                                   float* __restrict__ out) {
  __shared__ float  Ebuf[16 * 32];      // full-H energy tile, cross-wave reduced
  __shared__ __bf16 Ktile[32 * 512];    // 32 KB staged K block (shared by 8 waves)

  const int wave = threadIdx.x >> 5;
  const int lane = threadIdx.x & 31;
  const int hi   = lane >> 4;
  const int l15  = lane & 15;
  const int b    = blockIdx.x >> 7;
  const int t0   = (blockIdx.x & 127) << 4;
  const int hbase = wave * 64;

  // Preload the two Q A-fragments (h0 = hbase, hbase+32) — reused every key block
  const __bf16* qrow = Qp + ((size_t)(b * 2048 + t0 + l15)) * 512;
  v16bf aq[2];
#pragma unroll
  for (int j = 0; j < 2; ++j) {
    int h0 = hbase + 32 * j;
    v8bf g1 = *(const v8bf*)(qrow + h0 + hi * 8);
    v8bf g2 = *(const v8bf*)(qrow + h0 + 16 + hi * 8);
#pragma unroll
    for (int i = 0; i < 8; ++i) { aq[j][i] = g1[i]; aq[j][i + 8] = g2[i]; }
  }

  float m_r = -1e30f, l_r = 0.0f;     // online softmax state (row = lane&15)
  v8f acc[4];
#pragma unroll
  for (int j = 0; j < 4; ++j)
#pragma unroll
    for (int i = 0; i < 8; ++i) acc[j][i] = 0.0f;

  // Issue TDM for the first K block
  if (wave == 0)
    tdm_load_kblock(Kp + ((size_t)(b * 2048)) * 512, Ktile);

  for (int s0 = 0; s0 < 2048; s0 += 32) {
    // ---- prefetch next value block into cache (global_prefetch_b8) ----
    if (s0 + 32 < 2048) {
      const __bf16* pv =
          Vt + ((size_t)(b * 512 + hbase + (lane & 3) * 16 + (lane >> 2))) * 2048 +
          s0 + 32;
      __builtin_prefetch(pv, 0, 1);
    }

    // ---- wave 0 waits for the staged K block; barrier publishes it ----
    if (wave == 0) __builtin_amdgcn_s_wait_tensorcnt(0);
    // ---- zero shared energy tile ----
    if (threadIdx.x < 128) ((float4*)Ebuf)[threadIdx.x] = make_float4(0.f, 0.f, 0.f, 0.f);
    __syncthreads();

    // ---- partial energy for this wave's H-chunk: two 16x16 s-halves ----
#pragma unroll
    for (int sh = 0; sh < 32; sh += 16) {
      v8f e = {0.f, 0.f, 0.f, 0.f, 0.f, 0.f, 0.f, 0.f};
      const __bf16* krow = &Ktile[(sh + l15) * 512 + hbase + hi * 8];
#pragma unroll
      for (int j = 0; j < 2; ++j) {
        v8bf b1 = *(const v8bf*)(krow + 32 * j);
        v8bf b2 = *(const v8bf*)(krow + 32 * j + 16);
        v16bf bk;
#pragma unroll
        for (int i = 0; i < 8; ++i) { bk[i] = b1[i]; bk[i + 8] = b2[i]; }
        e = __builtin_amdgcn_wmma_f32_16x16x32_bf16(false, aq[j], false, bk,
                                                    (short)0, e, false, false);
      }
#pragma unroll
      for (int i = 0; i < 8; ++i)
        atomicAdd(&Ebuf[(i + hi * 8) * 32 + sh + l15], e[i]);
    }
    __syncthreads();   // all waves done reading Ktile + Ebuf complete

    // ---- kick off TDM for the next K block (runs under softmax + PV) ----
    if (wave == 0 && s0 + 32 < 2048)
      tdm_load_kblock(Kp + ((size_t)(b * 2048 + s0 + 32)) * 512, Ktile);

    // ---- online softmax update (replicated per wave; lane pair shares a row) ----
    float ev[16];
#pragma unroll
    for (int i = 0; i < 8; ++i) {
      ev[i]     = Ebuf[l15 * 32 + hi * 16 + i]      * SCALE_F;
      ev[i + 8] = Ebuf[l15 * 32 + hi * 16 + 8 + i] * SCALE_F;
    }
    float mx = ev[0];
#pragma unroll
    for (int i = 1; i < 16; ++i) mx = fmaxf(mx, ev[i]);
    mx = fmaxf(mx, __shfl_xor(mx, 16, 32));
    const float m_new = fmaxf(m_r, mx);
    float psum = 0.0f;
    v16bf pf;                         // P already lands in A-fragment layout
#pragma unroll
    for (int i = 0; i < 16; ++i) {
      float p = __expf(ev[i] - m_new);
      psum += p;
      pf[i] = (__bf16)p;
    }
    psum += __shfl_xor(psum, 16, 32);
    const float alpha = __expf(m_r - m_new);
    l_r = l_r * alpha + psum;
    m_r = m_new;
    __syncthreads();                  // Ebuf reads done before next-iter zeroing

    // ---- rescale accumulators and accumulate P x V for this H-chunk ----
#pragma unroll
    for (int i = 0; i < 8; ++i) {
      float ai = __shfl(alpha, i + hi * 8, 32);
#pragma unroll
      for (int j = 0; j < 4; ++j) acc[j][i] *= ai;
    }
#pragma unroll
    for (int j = 0; j < 4; ++j) {
      const __bf16* vrow =
          Vt + ((size_t)(b * 512 + hbase + j * 16 + l15)) * 2048 + s0 + hi * 16;
      v8bf b1 = *(const v8bf*)vrow;
      v8bf b2 = *(const v8bf*)(vrow + 8);
      v16bf bv;
#pragma unroll
      for (int i = 0; i < 8; ++i) { bv[i] = b1[i]; bv[i + 8] = b2[i]; }
      acc[j] = __builtin_amdgcn_wmma_f32_16x16x32_bf16(false, pf, false, bv,
                                                       (short)0, acc[j], false, false);
    }
  }

  // ---- finalize: divide by softmax denominator, store fp32 output ----
#pragma unroll
  for (int i = 0; i < 8; ++i) {
    float li  = __shfl(l_r, i + hi * 8, 32);
    float inv = 1.0f / li;
    int t = t0 + i + hi * 8;
#pragma unroll
    for (int j = 0; j < 4; ++j)
      out[((size_t)(b * 2048 + t)) * 512 + hbase + j * 16 + l15] = acc[j][i] * inv;
  }
}

// ---------------------------------------------------------------------------
extern "C" void kernel_launch(void* const* d_in, const int* in_sizes, int n_in,
                              void* d_out, int out_size, void* d_ws, size_t ws_size,
                              hipStream_t stream) {
  (void)in_sizes; (void)n_in; (void)out_size; (void)ws_size;
  const float* q  = (const float*)d_in[0];
  const float* k  = (const float*)d_in[1];
  const float* v  = (const float*)d_in[2];
  const float* Wq = (const float*)d_in[3];
  const float* bq = (const float*)d_in[4];
  const float* Wk = (const float*)d_in[5];
  const float* bk = (const float*)d_in[6];
  const float* Wv = (const float*)d_in[7];
  const float* bv = (const float*)d_in[8];
  float* out = (float*)d_out;

  // Workspace layout (bf16 elements): Qp | Kp | Vt | Wt  (~51.9 MB total)
  __bf16* Qp = (__bf16*)d_ws;
  __bf16* Kp = Qp + (size_t)16384 * 512;
  __bf16* Vt = Kp + (size_t)16384 * 512;
  __bf16* Wt = Vt + (size_t)8 * 512 * 2048;

  wtrans_kernel<<<(3 * 512 * 512) / 256, 256, 0, stream>>>(Wq, Wk, Wv, Wt);
  // 3 proj * 1024 m-tiles * 8 n-groups = 24576 waves, 8 waves/block
  qkv_proj_kernel<<<24576 / 8, 256, 0, stream>>>(q, k, v, bq, bk, bv, Wt, Qp, Kp, Vt);
  // 8 batches * 128 query tiles
  attn_kernel<<<8 * 128, 256, 0, stream>>>(Qp, Kp, Vt, out);
}
